// PCL_41051297415824
// MI455X (gfx1250) — matrix operand order repaired and checked by reading, not compiled, not run on hardware
//
#include <hip/hip_runtime.h>
#include <hip/hip_bf16.h>

#define NPTS 16384
#define DIMS 512
#define KCL  1024
#define KM_ITERS 10
#define CONC_INV 10.0f   /* 1/CONCENTRATION */
#define TEMP_INV 10.0f   /* 1/TEMPERATURE   */
#define EPSV 1e-6f

typedef __attribute__((ext_vector_type(16))) __bf16 v16bf;
typedef __attribute__((ext_vector_type(8)))  __bf16 v8bf;
typedef __attribute__((ext_vector_type(8)))  float  v8f;

union BFV { v16bf v; v8bf h[2]; };

// ---------------- feats f32 -> bf16 hi/lo split ----------------
__global__ __launch_bounds__(256) void k_prep_feats(const float* __restrict__ f,
                                                    unsigned short* __restrict__ fh_u,
                                                    unsigned short* __restrict__ fl_u) {
  __bf16* fh = (__bf16*)fh_u;
  __bf16* fl = (__bf16*)fl_u;
  long i = (long)blockIdx.x * blockDim.x + threadIdx.x;
  if (i >= (long)NPTS * DIMS) return;
  float v = f[i];
  __bf16 hi = (__bf16)v;
  fh[i] = hi;
  fl[i] = (__bf16)(v - (float)hi);
}

// ---------------- centroids init: feats[:K] ----------------
__global__ __launch_bounds__(256) void k_init_cent(const float* __restrict__ f,
                                                   float* __restrict__ cent) {
  long i = (long)blockIdx.x * blockDim.x + threadIdx.x;
  if (i >= (long)KCL * DIMS) return;
  cent[i] = f[i];
}

// ---------------- per-iteration: centroids -> bf16 hi/lo + c_sq ----------------
__global__ __launch_bounds__(256) void k_cent_prep(const float* __restrict__ cent,
                                                   unsigned short* __restrict__ ch_u,
                                                   unsigned short* __restrict__ cl_u,
                                                   float* __restrict__ c_sq) {
  __bf16* ch = (__bf16*)ch_u;
  __bf16* cl = (__bf16*)cl_u;
  const int c = blockIdx.x;      // one block per cluster
  const int t = threadIdx.x;     // 256 threads, DIMS=512 -> 2 elems each
  float ssq = 0.f;
  for (int d = t; d < DIMS; d += 256) {
    float v = cent[(size_t)c * DIMS + d];
    __bf16 hi = (__bf16)v;
    ch[(size_t)c * DIMS + d] = hi;
    cl[(size_t)c * DIMS + d] = (__bf16)(v - (float)hi);
    ssq += v * v;
  }
  __shared__ float red[256];
  red[t] = ssq;
  __syncthreads();
  for (int s = 128; s > 0; s >>= 1) {
    if (t < s) red[t] += red[t + s];
    __syncthreads();
  }
  if (t == 0) c_sq[c] = red[0];
}

// ---------------- WMMA assignment: argmin_k (c_sq[k] - 2*dot) ----------------
// One wave handles 16 rows x all 1024 columns. 3 bf16 WMMAs per K=32 chunk
// (hi*hi + hi*lo + lo*hi) for near-fp32 dot products.
__global__ __launch_bounds__(256) void k_assign(const unsigned short* __restrict__ fh_u,
                                                const unsigned short* __restrict__ fl_u,
                                                const unsigned short* __restrict__ ch_u,
                                                const unsigned short* __restrict__ cl_u,
                                                const float* __restrict__ c_sq,
                                                int* __restrict__ clusters) {
  const __bf16* fh = (const __bf16*)fh_u;
  const __bf16* fl = (const __bf16*)fl_u;
  const __bf16* ch = (const __bf16*)ch_u;
  const __bf16* cl = (const __bf16*)cl_u;

  const int lane = threadIdx.x & 31;
  const int wave = blockIdx.x * (blockDim.x >> 5) + (threadIdx.x >> 5);
  const int m0   = wave * 16;            // 16-row band
  const int nrow = lane & 15;            // A: M = lane&15 ; B/C: N = lane&15
  const int half = lane >> 4;            // upper/lower half-wave
  const int kA   = half * 8;             // A K-group offset (ISA layout)
  const int kB   = half * 8;             // B K-group offset (ISA layout)

  const __bf16* ahp = fh + (size_t)(m0 + nrow) * DIMS + kA;
  const __bf16* alp = fl + (size_t)(m0 + nrow) * DIMS + kA;

  float best[8];
  int   bidx[8];
#pragma unroll
  for (int r = 0; r < 8; ++r) { best[r] = 3.4e38f; bidx[r] = 0; }

  for (int ct = 0; ct < KCL / 16; ++ct) {
    const int ccol = ct * 16 + nrow;     // this lane's column (cluster index)
    const __bf16* bhp = ch + (size_t)ccol * DIMS + kB;
    const __bf16* blp = cl + (size_t)ccol * DIMS + kB;
    v8f acc = {};
    for (int kc = 0; kc < DIMS; kc += 32) {
      BFV ah, al, bh, bl;
      ah.h[0] = *(const v8bf*)(ahp + kc);
      ah.h[1] = *(const v8bf*)(ahp + kc + 16);
      al.h[0] = *(const v8bf*)(alp + kc);
      al.h[1] = *(const v8bf*)(alp + kc + 16);
      bh.h[0] = *(const v8bf*)(bhp + kc);
      bh.h[1] = *(const v8bf*)(bhp + kc + 16);
      bl.h[0] = *(const v8bf*)(blp + kc);
      bl.h[1] = *(const v8bf*)(blp + kc + 16);
      acc = __builtin_amdgcn_wmma_f32_16x16x32_bf16(false, ah.v, false, bh.v,
                                                    (short)0, acc, false, false);
      acc = __builtin_amdgcn_wmma_f32_16x16x32_bf16(false, ah.v, false, bl.v,
                                                    (short)0, acc, false, false);
      acc = __builtin_amdgcn_wmma_f32_16x16x32_bf16(false, al.v, false, bh.v,
                                                    (short)0, acc, false, false);
    }
    const float csq = c_sq[ccol];
#pragma unroll
    for (int r = 0; r < 8; ++r) {
      float d = csq - 2.0f * acc[r];     // row M = m0 + r + 8*half, col = ccol
      if (d < best[r] || (d == best[r] && ccol < bidx[r])) { best[r] = d; bidx[r] = ccol; }
    }
  }
  // reduce min/argmin across the 16 lanes that share each row (width-16 xor)
#pragma unroll
  for (int off = 8; off >= 1; off >>= 1) {
#pragma unroll
    for (int r = 0; r < 8; ++r) {
      float ob = __shfl_xor(best[r], off, 16);
      int   oi = __shfl_xor(bidx[r], off, 16);
      if (ob < best[r] || (ob == best[r] && oi < bidx[r])) { best[r] = ob; bidx[r] = oi; }
    }
  }
  if (nrow == 0) {
#pragma unroll
    for (int r = 0; r < 8; ++r)
      clusters[m0 + 8 * half + r] = bidx[r];
  }
}

// ---------------- generic zero ----------------
__global__ __launch_bounds__(256) void k_zero(float* __restrict__ p, long n) {
  long i = (long)blockIdx.x * blockDim.x + threadIdx.x;
  if (i < n) p[i] = 0.f;
}

// ---------------- segment sums (atomic) ----------------
__global__ __launch_bounds__(256) void k_accum(const float* __restrict__ feats,
                                               const int* __restrict__ clusters,
                                               float* __restrict__ sums,
                                               float* __restrict__ counts) {
  long idx = (long)blockIdx.x * blockDim.x + threadIdx.x;   // N * D/4 threads
  if (idx >= (long)NPTS * (DIMS / 4)) return;
  int n  = (int)(idx >> 7);
  int d4 = ((int)idx & 127) << 2;
  int c  = clusters[n];
  float4 v = *(const float4*)(feats + (size_t)n * DIMS + d4);
  float* s = sums + (size_t)c * DIMS + d4;
  atomicAdd(s + 0, v.x);
  atomicAdd(s + 1, v.y);
  atomicAdd(s + 2, v.z);
  atomicAdd(s + 3, v.w);
  if (d4 == 0) atomicAdd(&counts[c], 1.0f);
}

// ---------------- centroid update ----------------
__global__ __launch_bounds__(256) void k_update(float* __restrict__ cent,
                                                const float* __restrict__ sums,
                                                const float* __restrict__ counts) {
  long i = (long)blockIdx.x * blockDim.x + threadIdx.x;     // K*D threads
  if (i >= (long)KCL * DIMS) return;
  float cnt = counts[i >> 9];
  if (cnt > 0.f) cent[i] = sums[i] / cnt;   // counts>0 => >=1, max(cnt,1)==cnt
}

// ---------------- residual^2 segment sum ----------------
__global__ __launch_bounds__(256) void k_resid(const float* __restrict__ feats,
                                               const int* __restrict__ clusters,
                                               const float* __restrict__ cent,
                                               float* __restrict__ dists) {
  long idx = (long)blockIdx.x * blockDim.x + threadIdx.x;   // N * D/4 threads
  if (idx >= (long)NPTS * (DIMS / 4)) return;
  int n  = (int)(idx >> 7);
  int d4 = ((int)idx & 127) << 2;
  int c  = clusters[n];
  float4 f = *(const float4*)(feats + (size_t)n * DIMS + d4);
  float4 m = *(const float4*)(cent + (size_t)c * DIMS + d4);
  float* dd = dists + (size_t)c * DIMS + d4;
  float rx = f.x - m.x, ry = f.y - m.y, rz = f.z - m.z, rw = f.w - m.w;
  atomicAdd(dd + 0, rx * rx);
  atomicAdd(dd + 1, ry * ry);
  atomicAdd(dd + 2, rz * rz);
  atomicAdd(dd + 3, rw * rw);
}

// ---------------- reductions ----------------
__global__ __launch_bounds__(256) void k_red1(const float* __restrict__ dists,
                                              float* __restrict__ sumP) {
  __shared__ float red[256];
  long stride = (long)gridDim.x * blockDim.x;
  float s = 0.f;
  for (long i = (long)blockIdx.x * blockDim.x + threadIdx.x; i < (long)KCL * DIMS; i += stride)
    s += expf(-dists[i] * CONC_INV);
  red[threadIdx.x] = s;
  __syncthreads();
  for (int t = 128; t > 0; t >>= 1) {
    if ((int)threadIdx.x < t) red[threadIdx.x] += red[threadIdx.x + t];
    __syncthreads();
  }
  if (threadIdx.x == 0) atomicAdd(sumP, red[0]);
}

__global__ __launch_bounds__(256) void k_red2(const float* __restrict__ dists,
                                              const float* __restrict__ sumP,
                                              float* __restrict__ acc) {
  __shared__ float red[256];
  const float S = sumP[0];
  long stride = (long)gridDim.x * blockDim.x;
  float s = 0.f;
  for (long i = (long)blockIdx.x * blockDim.x + threadIdx.x; i < (long)KCL * DIMS; i += stride) {
    float d  = dists[i];
    float p  = expf(-d * CONC_INV);
    float Q  = p / S;
    float p2 = expf(-d * TEMP_INV);
    s += Q * logf(Q + EPSV) - logf(p2 + EPSV);   // entropy term - log p2 term
  }
  red[threadIdx.x] = s;
  __syncthreads();
  for (int t = 128; t > 0; t >>= 1) {
    if ((int)threadIdx.x < t) red[threadIdx.x] += red[threadIdx.x + t];
    __syncthreads();
  }
  if (threadIdx.x == 0) atomicAdd(acc, red[0]);
}

__global__ void k_fin(const float* __restrict__ acc, float* __restrict__ out) {
  out[0] = acc[0] * (1.0f / ((float)KCL * (float)DIMS));
}

// ---------------- host orchestration ----------------
extern "C" void kernel_launch(void* const* d_in, const int* in_sizes, int n_in,
                              void* d_out, int out_size, void* d_ws, size_t ws_size,
                              hipStream_t stream) {
  const float* feats = (const float*)d_in[0];
  float* out = (float*)d_out;

  char* p = (char*)d_ws;
  auto alloc = [&](size_t bytes) -> void* {
    void* r = (void*)p;
    p += (bytes + 255) & ~(size_t)255;
    return r;
  };
  unsigned short* fh   = (unsigned short*)alloc((size_t)NPTS * DIMS * 2); // feats hi bf16
  unsigned short* fl   = (unsigned short*)alloc((size_t)NPTS * DIMS * 2); // feats lo bf16
  float*          cent = (float*)alloc((size_t)KCL * DIMS * 4);
  unsigned short* chB  = (unsigned short*)alloc((size_t)KCL * DIMS * 2);
  unsigned short* clB  = (unsigned short*)alloc((size_t)KCL * DIMS * 2);
  float*          csq  = (float*)alloc((size_t)KCL * 4);
  int*            clus = (int*)alloc((size_t)NPTS * 4);
  float*          sums = (float*)alloc((size_t)KCL * DIMS * 4);
  float*          cnts = (float*)alloc((size_t)KCL * 4);
  float*          dst2 = (float*)alloc((size_t)KCL * DIMS * 4);
  float*          scal = (float*)alloc(2 * 4);   // [0]=sumP  [1]=final accumulator

  const long ND  = (long)NPTS * DIMS;      // 8388608
  const long KD  = (long)KCL * DIMS;       // 524288
  const long NQ  = (long)NPTS * (DIMS / 4);

  k_prep_feats<<<(int)((ND + 255) / 256), 256, 0, stream>>>(feats, fh, fl);
  k_init_cent<<<(int)((KD + 255) / 256), 256, 0, stream>>>(feats, cent);

  for (int it = 0; it < KM_ITERS; ++it) {
    k_cent_prep<<<KCL, 256, 0, stream>>>(cent, chB, clB, csq);
    k_assign<<<NPTS / 16 / 8, 256, 0, stream>>>(fh, fl, chB, clB, csq, clus);
    k_zero<<<(int)((KD + 255) / 256), 256, 0, stream>>>(sums, KD);
    k_zero<<<(KCL + 255) / 256, 256, 0, stream>>>(cnts, KCL);
    k_accum<<<(int)((NQ + 255) / 256), 256, 0, stream>>>(feats, clus, sums, cnts);
    k_update<<<(int)((KD + 255) / 256), 256, 0, stream>>>(cent, sums, cnts);
  }

  // final assignment with converged centroids
  k_cent_prep<<<KCL, 256, 0, stream>>>(cent, chB, clB, csq);
  k_assign<<<NPTS / 16 / 8, 256, 0, stream>>>(fh, fl, chB, clB, csq, clus);

  // per-cluster residual sums and scalar loss
  k_zero<<<(int)((KD + 255) / 256), 256, 0, stream>>>(dst2, KD);
  k_resid<<<(int)((NQ + 255) / 256), 256, 0, stream>>>(feats, clus, cent, dst2);
  k_zero<<<1, 256, 0, stream>>>(scal, 2);
  k_red1<<<512, 256, 0, stream>>>(dst2, &scal[0]);
  k_red2<<<512, 256, 0, stream>>>(dst2, &scal[0], &scal[1]);
  k_fin<<<1, 1, 0, stream>>>(&scal[1], out);
}
